// Hyper_Fusion_25924422598802
// MI455X (gfx1250) — matrix operand order repaired and checked by reading, not compiled, not run on hardware
//
#include <hip/hip_runtime.h>

// ---------------------------------------------------------------------------
// MI455X (gfx1250) fused cross-attention, bf16 WMMA path + TDM K-tile staging.
//   M = C = 256, B = 4, HW = 4096, softmax over BATCH axis (torch dim=0 quirk)
// ---------------------------------------------------------------------------

typedef __bf16 bf16_t;
typedef __attribute__((ext_vector_type(16))) __bf16 v16bf;
typedef __attribute__((ext_vector_type(8)))  __bf16 v8bf;
typedef __attribute__((ext_vector_type(8)))  float  v8f;
typedef __attribute__((ext_vector_type(4)))  unsigned int u32x4;
typedef __attribute__((ext_vector_type(8)))  int i32x8;
typedef __attribute__((ext_vector_type(4)))  int i32x4;

#if defined(__has_builtin)
#  if __has_builtin(__builtin_amdgcn_tensor_load_to_lds)
#    define HAVE_TDM 1
#  endif
#endif
#ifndef HAVE_TDM
#  define HAVE_TDM 0
#endif

union FragBF { v16bf v; v8bf h[2]; };

__device__ __forceinline__ v8f v8f_zero() {
    v8f z;
#pragma unroll
    for (int i = 0; i < 8; ++i) z[i] = 0.0f;
    return z;
}

__device__ __forceinline__ v8f wmma_bf16(v16bf a, v16bf b, v8f c) {
    // D = A(16x32 bf16) x B(32x16 bf16) + C(16x16 f32)
    return __builtin_amdgcn_wmma_f32_16x16x32_bf16(
        /*neg_a=*/false, a, /*neg_b=*/false, b,
        /*c_mod=*/(short)0, c, /*reuse_a=*/false, /*reuse_b=*/false);
}

#if HAVE_TDM
// TDM: DMA one contiguous 16 KB tile (8192 bf16) from global into LDS.
// D# built per CDNA5 ISA:
//   group0: count=1 | lds_addr | global_addr[56:0] | type=2
//   group1: data_size=2B; tensor_dim0 = tile_dim0 = 8192; tensor_dim1 = 1;
//           tensor_dim0_stride = 8192. groups 2/3 (and the trailing 8-dword
//           group of the 6-arg clang-23 builtin) zero: tensor is <= 2D.
__device__ __forceinline__ void tdm_load_tile16k(const bf16_t* gsrc, unsigned int lds_off) {
    unsigned long long ga = (unsigned long long)gsrc;
    u32x4 g0; i32x8 g1; i32x4 g2; i32x4 g3; i32x8 g4;
    g0[0] = 1u;                                        // count = 1 (valid D#)
    g0[1] = lds_off;                                   // lds_addr (bytes)
    g0[2] = (unsigned int)ga;                          // global_addr[31:0]
    g0[3] = (unsigned int)(ga >> 32) | (2u << 30);     // global_addr[56:32] | type=2
    g1[0] = 0x00010000;                                // data_size = 1 -> 2 bytes
    g1[1] = (int)(8192u << 16);                        // tensor_dim0[15:0] << 16
    g1[2] = (int)((8192u >> 16) | (1u << 16));         // tensor_dim0[31:16] | tensor_dim1 = 1
    g1[3] = (int)(8192u << 16);                        // tensor_dim1[31:16]=0 | tile_dim0 = 8192
    g1[4] = 1;                                         // tile_dim1 = 1, tile_dim2 = 0
    g1[5] = 8192;                                      // tensor_dim0_stride[31:0]
    g1[6] = 0; g1[7] = 0;
    g2[0] = g2[1] = g2[2] = g2[3] = 0;
    g3[0] = g3[1] = g3[2] = g3[3] = 0;
#pragma unroll
    for (int i = 0; i < 8; ++i) g4[i] = 0;
    __builtin_amdgcn_tensor_load_to_lds(g0, g1, g2, g3, g4, 0);
}
#endif

// ---------------------------------------------------------------------------
// Kernel 1: f32 weights -> bf16, TRANSPOSED: Wt[o][n_out][k_in] = W_o[k_in][n_out]
// ---------------------------------------------------------------------------
__global__ __launch_bounds__(256) void k_cvt_w(
    const float* __restrict__ Wq, const float* __restrict__ Wk,
    const float* __restrict__ Wv, const float* __restrict__ Wo,
    bf16_t* __restrict__ Wt) {
    int o = blockIdx.y;
    const float* W = (o == 0) ? Wq : (o == 1) ? Wk : (o == 2) ? Wv : Wo;
    int idx = blockIdx.x * 256 + threadIdx.x;   // 0..65535
    int n = idx >> 8, k = idx & 255;
    Wt[o * 65536 + n * 256 + k] = (bf16_t)W[k * 256 + n];
}

// ---------------------------------------------------------------------------
// Kernel 2: LayerNorm over C=256; one wave per (side,b,n) row; shfl reduce.
// ---------------------------------------------------------------------------
__global__ __launch_bounds__(256) void k_ln(
    const float* __restrict__ xl, const float* __restrict__ xr,
    const float* __restrict__ g1, const float* __restrict__ b1,
    const float* __restrict__ g2, const float* __restrict__ b2,
    bf16_t* __restrict__ xln) {
    int row  = blockIdx.x * 8 + (threadIdx.x >> 5);   // 0..32767
    int lane = threadIdx.x & 31;
    int side = row >> 14;
    int r    = row & 16383;
    int b    = r >> 12, n = r & 4095;
    const float* x  = side ? xr : xl;
    const float* g  = side ? g2 : g1;
    const float* be = side ? b2 : b1;
    const float* base = x + (size_t)b * 256 * 4096 + n;

    float v[8], s = 0.0f, q = 0.0f;
#pragma unroll
    for (int j = 0; j < 8; ++j) {
        int c = lane + 32 * j;
        float t = base[(size_t)c * 4096];
        v[j] = t; s += t; q += t * t;
    }
#pragma unroll
    for (int k = 16; k > 0; k >>= 1) {
        s += __shfl_xor(s, k, 32);
        q += __shfl_xor(q, k, 32);
    }
    float mu  = s * (1.0f / 256.0f);
    float var = q * (1.0f / 256.0f) - mu * mu;   // biased var (matches jnp.var)
    float rs  = rsqrtf(var + 1e-5f);

    bf16_t* o = xln + (size_t)row * 256;
#pragma unroll
    for (int j = 0; j < 8; ++j) {
        int c = lane + 32 * j;
        o[c] = (bf16_t)((v[j] - mu) * rs * g[c] + be[c]);
    }
}

// ---------------------------------------------------------------------------
// Kernel 3: QKV projection GEMM (bf16 WMMA). V stored transposed -> Vt[d][m].
// ---------------------------------------------------------------------------
__global__ __launch_bounds__(256) void k_qkv(
    const bf16_t* __restrict__ xln, const bf16_t* __restrict__ Wt,
    bf16_t* __restrict__ Q, bf16_t* __restrict__ K, bf16_t* __restrict__ Vt) {
    int o    = blockIdx.y;              // 0:Q 1:K 2:V
    int rt   = blockIdx.x;              // 16-row tile, 0..2047
    int w    = threadIdx.x >> 5;
    int lane = threadIdx.x & 31;
    int lh   = lane >> 4;
    int l16  = lane & 15;

    const bf16_t* A = xln + (size_t)(rt * 16 + l16) * 256;
    const bf16_t* B = Wt + o * 65536;

    v8f acc0 = v8f_zero(), acc1 = v8f_zero();
#pragma unroll
    for (int ks = 0; ks < 8; ++ks) {
        FragBF a;
        a.h[0] = *(const v8bf*)(A + ks * 32 + lh * 16);
        a.h[1] = *(const v8bf*)(A + ks * 32 + lh * 16 + 8);
        v16bf b0 = *(const v16bf*)(B + (size_t)(w * 32 + l16) * 256      + ks * 32 + lh * 16);
        v16bf b1 = *(const v16bf*)(B + (size_t)(w * 32 + 16 + l16) * 256 + ks * 32 + lh * 16);
        acc0 = wmma_bf16(a.v, b0, acc0);
        acc1 = wmma_bf16(a.v, b1, acc1);
    }

    if (o < 2) {
        bf16_t* O = (o == 0) ? Q : K;
#pragma unroll
        for (int v = 0; v < 8; ++v) {
            int rr = rt * 16 + v + 8 * lh;
            O[(size_t)rr * 256 + w * 32 + l16]      = (bf16_t)acc0[v];
            O[(size_t)rr * 256 + w * 32 + 16 + l16] = (bf16_t)acc1[v];
        }
    } else {
#pragma unroll
        for (int v = 0; v < 8; ++v) {
            int rr = rt * 16 + v + 8 * lh;
            int sb = rr >> 12;          // side*4 + b
            int m  = rr & 4095;
            Vt[(size_t)sb * 256 * 4096 + (size_t)(w * 32 + l16) * 4096 + m]      = (bf16_t)acc0[v];
            Vt[(size_t)sb * 256 * 4096 + (size_t)(w * 32 + 16 + l16) * 4096 + m] = (bf16_t)acc1[v];
        }
    }
}

// ---------------------------------------------------------------------------
// Kernel 4: fused attention.
//   Block = 8 waves = (b, nsub). K tiles (32 m x 256 d, per batch) are staged
//   into LDS by the Tensor Data Mover, double-buffered: the DMA for tile m32+1
//   runs while the WMMAs for tile m32 execute; consumers read K via ds_load.
//   Softmax over the batch axis is elementwise in the WMMA C/D layout (tiny
//   LDS exchange of four 16x16 S tiles, no row reductions). V B-operands
//   stream from global/L2 on the vector pipe, overlapping the DS pipe.
// ---------------------------------------------------------------------------
__global__ __launch_bounds__(256) void k_attn(
    const bf16_t* __restrict__ Q, const bf16_t* __restrict__ Km,
    const bf16_t* __restrict__ Vt, bf16_t* __restrict__ Y) {
    __shared__ bf16_t sK[2][4][32 * 256];  // 128 KB double-buffered K tiles
    __shared__ float  sS[2][4][256];       // batch-softmax exchange (8 KB)
    __shared__ bf16_t sP[8][16 * 32];      // wave-private P tile, D->A bounce (8 KB)

    int side = blockIdx.y;
    int w    = threadIdx.x >> 5;
    int lane = threadIdx.x & 31;
    int b    = w & 3;
    int nsub = w >> 2;
    int lh   = lane >> 4;
    int l16  = lane & 15;
    int n0   = blockIdx.x * 32 + nsub * 16;

    const bf16_t* Qp = Q  + (size_t)(side * 4 + b) * 4096 * 256;
    const bf16_t* Vp = Vt + (size_t)(side * 4 + b) * 256 * 4096;
    // Issuing waves (w < 4) stage batch b = w of the *opposite* side's K.
    const bf16_t* Kissue = Km + (size_t)((1 - side) * 4 + w) * 4096 * 256;

#if !HAVE_TDM
    auto stage = [&](int buf, int t32) {   // cooperative fallback: 64 KB copy
        int t = threadIdx.x;
#pragma unroll
        for (int i = 0; i < 16; ++i) {
            int c   = t + i * 256;         // 4096 x 16B chunks
            int bb  = c >> 10;
            int off = (c & 1023) * 8;
            const bf16_t* s = Km + (size_t)((1 - side) * 4 + bb) * 4096 * 256
                                 + (size_t)t32 * 8192 + off;
            *(v8bf*)&sK[buf][bb][off] = *(const v8bf*)s;
        }
    };
#endif

    // Preload this wave's Q A-fragments (16 rows x 256 d) once.
    FragBF qf[8];
    const bf16_t* qrow = Qp + (size_t)(n0 + l16) * 256;
#pragma unroll
    for (int ks = 0; ks < 8; ++ks) {
        qf[ks].h[0] = *(const v8bf*)(qrow + ks * 32 + lh * 16);
        qf[ks].h[1] = *(const v8bf*)(qrow + ks * 32 + lh * 16 + 8);
    }

    v8f yacc[16];
#pragma unroll
    for (int d = 0; d < 16; ++d) yacc[d] = v8f_zero();

    const float scale = 0.0625f;   // 1/sqrt(256)

    // Prologue: stage K tiles for m32 = 0 into buffer 0.
#if HAVE_TDM
    if (w < 4)
        tdm_load_tile16k(Kissue, (unsigned int)(unsigned long long)&sK[0][w][0]);
#else
    stage(0, 0);
#endif

    for (int m32 = 0; m32 < 128; ++m32) {
        int cur = m32 & 1;
#if HAVE_TDM
        if (w < 4) __builtin_amdgcn_s_wait_tensorcnt(0);   // own DMA complete
#endif
        __syncthreads();   // publish K[cur]; previous buffer's readers are done
#if HAVE_TDM
        if (w < 4 && m32 + 1 < 128)
            tdm_load_tile16k(Kissue + (size_t)(m32 + 1) * 8192,
                             (unsigned int)(unsigned long long)&sK[cur ^ 1][w][0]);
#else
        if (m32 + 1 < 128) stage(cur ^ 1, m32 + 1);
#endif

#pragma unroll
        for (int sub = 0; sub < 2; ++sub) {
            v8f s = v8f_zero();
            const bf16_t* kt = &sK[cur][b][(sub * 16 + l16) * 256];
#pragma unroll
            for (int ks = 0; ks < 8; ++ks) {
                v16bf kb = *(const v16bf*)(kt + ks * 32 + lh * 16);
                s = wmma_bf16(qf[ks].v, kb, s);
            }
            __syncthreads();   // previous sS consumers done
#pragma unroll
            for (int v = 0; v < 8; ++v) sS[nsub][b][v * 32 + lane] = s[v];
            __syncthreads();   // all 4 batches' S tiles visible
#pragma unroll
            for (int v = 0; v < 8; ++v) {
                float t0 = sS[nsub][0][v * 32 + lane] * scale;
                float t1 = sS[nsub][1][v * 32 + lane] * scale;
                float t2 = sS[nsub][2][v * 32 + lane] * scale;
                float t3 = sS[nsub][3][v * 32 + lane] * scale;
                float mx = fmaxf(fmaxf(t0, t1), fmaxf(t2, t3));
                float e0 = __expf(t0 - mx), e1 = __expf(t1 - mx);
                float e2 = __expf(t2 - mx), e3 = __expf(t3 - mx);
                float inv = 1.0f / (e0 + e1 + e2 + e3);
                float eb  = (b == 0) ? e0 : (b == 1) ? e1 : (b == 2) ? e2 : e3;
                // D-layout (n = v + 8*lh, m = l16) -> row-major n x 32 in LDS
                sP[w][(v + 8 * lh) * 32 + sub * 16 + l16] = (bf16_t)(eb * inv);
            }
        }
        // Re-read P as a 16x32 A-operand fragment (wave-private LDS bounce).
        FragBF pf;
        {
            const bf16_t* pr = &sP[w][l16 * 32 + lh * 16];
            pf.h[0] = *(const v8bf*)(pr);
            pf.h[1] = *(const v8bf*)(pr + 8);
        }
        // Y += P x V over the 32-wide m slab, all 256 d columns (global/L2).
#pragma unroll
        for (int d = 0; d < 16; ++d) {
            const bf16_t* vptr = Vp + (size_t)(d * 16 + l16) * 4096 + m32 * 32 + lh * 16;
            v16bf vf = *(const v16bf*)vptr;
            if (m32 + 4 < 128) __builtin_prefetch(vptr + 128, 0, 0);  // pull-ahead
            yacc[d] = wmma_bf16(pf.v, vf, yacc[d]);
        }
    }

    bf16_t* Yp = Y + (size_t)(side * 4 + b) * 4096 * 256;
#pragma unroll
    for (int d = 0; d < 16; ++d)
#pragma unroll
        for (int v = 0; v < 8; ++v)
            Yp[(size_t)(n0 + v + 8 * lh) * 256 + d * 16 + l16] = (bf16_t)yacc[d][v];
}

// ---------------------------------------------------------------------------
// Kernel 5: out = x + Y @ Wo, written transposed back to (side, b, c, hw) f32.
// ---------------------------------------------------------------------------
__global__ __launch_bounds__(256) void k_out(
    const bf16_t* __restrict__ Y, const bf16_t* __restrict__ Wt,
    const float* __restrict__ xl, const float* __restrict__ xr,
    float* __restrict__ out) {
    int rt   = blockIdx.x;
    int w    = threadIdx.x >> 5;
    int lane = threadIdx.x & 31;
    int lh   = lane >> 4;
    int l16  = lane & 15;

    const bf16_t* A = Y + (size_t)(rt * 16 + l16) * 256;
    const bf16_t* B = Wt + 3 * 65536;   // Wo^T

    v8f acc0 = v8f_zero(), acc1 = v8f_zero();
#pragma unroll
    for (int ks = 0; ks < 8; ++ks) {
        FragBF a;
        a.h[0] = *(const v8bf*)(A + ks * 32 + lh * 16);
        a.h[1] = *(const v8bf*)(A + ks * 32 + lh * 16 + 8);
        v16bf b0 = *(const v16bf*)(B + (size_t)(w * 32 + l16) * 256      + ks * 32 + lh * 16);
        v16bf b1 = *(const v16bf*)(B + (size_t)(w * 32 + 16 + l16) * 256 + ks * 32 + lh * 16);
        acc0 = wmma_bf16(a.v, b0, acc0);
        acc1 = wmma_bf16(a.v, b1, acc1);
    }
#pragma unroll
    for (int v = 0; v < 8; ++v) {
        int rr   = rt * 16 + v + 8 * lh;
        int side = rr >> 14;
        int r2   = rr & 16383;
        int bb   = r2 >> 12, n = r2 & 4095;
        const float* x = side ? xr : xl;
        int c0 = w * 32 + l16, c1 = c0 + 16;
        size_t o0 = ((size_t)(side * 4 + bb) * 256 + c0) * 4096 + n;
        size_t o1 = ((size_t)(side * 4 + bb) * 256 + c1) * 4096 + n;
        size_t x0 = ((size_t)bb * 256 + c0) * 4096 + n;
        size_t x1 = ((size_t)bb * 256 + c1) * 4096 + n;
        out[o0] = x[x0] + acc0[v];
        out[o1] = x[x1] + acc1[v];
    }
}

// ---------------------------------------------------------------------------
extern "C" void kernel_launch(void* const* d_in, const int* in_sizes, int n_in,
                              void* d_out, int out_size, void* d_ws, size_t ws_size,
                              hipStream_t stream) {
    (void)in_sizes; (void)n_in; (void)out_size; (void)ws_size;
    const float* xl = (const float*)d_in[0];
    const float* xr = (const float*)d_in[1];
    const float* Wq = (const float*)d_in[2];
    const float* Wk = (const float*)d_in[3];
    const float* Wv = (const float*)d_in[4];
    const float* Wo = (const float*)d_in[5];
    const float* g1 = (const float*)d_in[6];
    const float* b1 = (const float*)d_in[7];
    const float* g2 = (const float*)d_in[8];
    const float* b2 = (const float*)d_in[9];
    float* out = (float*)d_out;

    char* ws = (char*)d_ws;
    const size_t SZ = (size_t)2 * 4 * 4096 * 256 * sizeof(bf16_t);  // 16 MiB
    bf16_t* xln = (bf16_t*)(ws);
    bf16_t* Wt  = (bf16_t*)(ws + SZ);                  // 512 KiB
    bf16_t* Q   = (bf16_t*)(ws + SZ + 524288);
    bf16_t* K   = (bf16_t*)(ws + 2 * SZ + 524288);
    bf16_t* Vt  = (bf16_t*)(ws + 3 * SZ + 524288);
    bf16_t* Y   = (bf16_t*)(ws + 4 * SZ + 524288);

    k_cvt_w<<<dim3(256, 4), 256, 0, stream>>>(Wq, Wk, Wv, Wo, Wt);
    k_ln   <<<4096,          256, 0, stream>>>(xl, xr, g1, b1, g2, b2, xln);
    k_qkv  <<<dim3(2048, 3), 256, 0, stream>>>(xln, Wt, Q, K, Vt);
    k_attn <<<dim3(128, 2),  256, 0, stream>>>(Q, K, Vt, Y);
    k_out  <<<2048,          256, 0, stream>>>(Y, Wt, xl, xr, out);
}
